// NetworkGNN_87282325389571
// MI455X (gfx1250) — matrix operand order repaired and checked by reading, not compiled, not run on hardware
//
#include <hip/hip_runtime.h>
#include <hip/hip_bf16.h>
#include <math.h>

#define N_NODES 50000
#define N_EDGES 600000
#define F_INN   128
#define F_EDGE  32
#define HIDD    128
#define NHEAD   8
#define DHEAD   16
#define NFAULT  5
#define NBATCH  8
#define EALL    (N_EDGES + N_NODES)
#define EPS_BN  1e-5f

typedef __attribute__((ext_vector_type(16))) __bf16       v16bf;
typedef __attribute__((ext_vector_type(8)))  float        v8f;
typedef __attribute__((ext_vector_type(4)))  unsigned int u32x4;

__device__ inline unsigned short f2bf(float f) {
  unsigned u = __float_as_uint(f);
  unsigned r = 0x7FFFu + ((u >> 16) & 1u);   // round-to-nearest-even
  return (unsigned short)((u + r) >> 16);
}

// float atomic max via signed/unsigned int ordering trick (init must be -inf)
__device__ inline void atomicMaxF(float* a, float v) {
  if (v >= 0.f) atomicMax((int*)a, __float_as_int(v));
  else          atomicMin((unsigned int*)a, __float_as_uint(v));
}

// ---------------- conversion / init ----------------

__global__ void k_f32_to_bf16(const float* __restrict__ src,
                              unsigned short* __restrict__ dst, int n) {
  int t = blockIdx.x * blockDim.x + threadIdx.x;
  if (t < n) dst[t] = f2bf(src[t]);
}

__global__ void k_fill_u32(unsigned int* __restrict__ p, int n, unsigned int v) {
  int t = blockIdx.x * blockDim.x + threadIdx.x;
  if (t < n) p[t] = v;
}

// ---------------- self-loop edge attr (mean of incoming) -----------

__global__ void k_loop_accum(const int* __restrict__ ei,
                             const float* __restrict__ eattr,
                             float* __restrict__ lattr,
                             float* __restrict__ cnt) {
  int e = blockIdx.x * blockDim.x + threadIdx.x;
  if (e >= N_EDGES) return;
  int d = ei[N_EDGES + e];
  atomicAdd(&cnt[d], 1.f);
  const float* ea = eattr + (size_t)e * F_EDGE;
  #pragma unroll
  for (int f = 0; f < F_EDGE; ++f) atomicAdd(&lattr[d * F_EDGE + f], ea[f]);
}

__global__ void k_loop_div(float* __restrict__ lattr, const float* __restrict__ cnt) {
  int t = blockIdx.x * blockDim.x + threadIdx.x;
  if (t >= N_NODES * F_EDGE) return;
  lattr[t] /= fmaxf(cnt[t >> 5], 1.f);
}

// ---------------- per-layer weight prep ----------------

// Veff[f][h] = sum_d We[f, h*16+d] * att_edge[h,d]   (32x8)
__global__ void k_veff(const float* __restrict__ We, const float* __restrict__ atte,
                       float* __restrict__ veff) {
  int t = threadIdx.x;            // 256 threads: f=t>>3, h=t&7
  int f = t >> 3, h = t & 7;
  float s = 0.f;
  #pragma unroll
  for (int d = 0; d < DHEAD; ++d) s += We[f * HIDD + h * DHEAD + d] * atte[h * DHEAD + d];
  veff[f * NHEAD + h] = s;
}

// pack W[128,128] f32 row-major into WMMA B-fragment order, bf16 pairs
// Bp[((kt*128 + c)*2 + half)*8 + j] = pack(W[kt*32+half*16+2j, c], W[...+1, c])
__global__ void k_pack_w(const float* __restrict__ W, unsigned int* __restrict__ Bp) {
  int t = blockIdx.x * blockDim.x + threadIdx.x;
  if (t >= 8192) return;
  int j  = t & 7;
  int u  = (t >> 3) & 1;
  int c  = (t >> 4) & 127;
  int kt = t >> 11;
  int k0 = kt * 32 + u * 16 + 2 * j;
  unsigned lo = f2bf(W[k0 * HIDD + c]);
  unsigned hi = f2bf(W[(k0 + 1) * HIDD + c]);
  Bp[t] = lo | (hi << 16);
}

// ---------------- bf16 WMMA GEMM: C[M,128] = A[M,128] x W[128,128] ----------

__global__ __launch_bounds__(256) void k_gemm_bf16(
    const unsigned short* __restrict__ A,   // [M,128] bf16 row-major
    const unsigned int* __restrict__ Bp,    // packed B fragments
    float* __restrict__ C) {
  int lane = threadIdx.x & 31;
  int wv   = threadIdx.x >> 5;              // col tile 0..7
  int mt   = blockIdx.x;                    // row tile
  int r    = lane & 15;
  int half = lane >> 4;
  int row  = mt * 16 + r;
  int col  = wv * 16 + r;
  v8f acc = {0.f, 0.f, 0.f, 0.f, 0.f, 0.f, 0.f, 0.f};
  #pragma unroll
  for (int t = 0; t < 4; ++t) {
    int kk = t * 32;
    union { u32x4 u[2]; v16bf v; } av, bv;
    av.u[0] = *(const u32x4*)(A + row * HIDD + kk + half * 16);       // K=kk(+16)..+7
    av.u[1] = *(const u32x4*)(A + row * HIDD + kk + half * 16 + 8);   // K=kk+8(+16)..
    const u32x4* bp = (const u32x4*)(Bp + ((t * 128 + col) * 2 + half) * 8);
    bv.u[0] = bp[0]; bv.u[1] = bp[1];
    acc = __builtin_amdgcn_wmma_f32_16x16x32_bf16(false, av.v, false, bv.v,
                                                  (short)0, acc, false, false);
  }
  int rowBase = mt * 16 + half * 8;
  #pragma unroll
  for (int j = 0; j < 8; ++j) C[(rowBase + j) * HIDD + col] = acc[j];
}

// ---------------- attention ----------------

__global__ void k_asrc_adst(const float* __restrict__ xs,
                            const float* __restrict__ atts, const float* __restrict__ attd,
                            float* __restrict__ asrc, float* __restrict__ adst) {
  int t = blockIdx.x * blockDim.x + threadIdx.x;
  if (t >= N_NODES * NHEAD) return;
  int n = t >> 3, h = t & 7;
  float s = 0.f, d = 0.f;
  #pragma unroll
  for (int k = 0; k < DHEAD; ++k) {
    float x = xs[(size_t)n * HIDD + h * DHEAD + k];
    s += x * atts[h * DHEAD + k];
    d += x * attd[h * DHEAD + k];
  }
  asrc[t] = s;
  adst[t] = d;
}

__global__ void k_att_logits(const int* __restrict__ ei,
                             const float* __restrict__ eattr,
                             const float* __restrict__ lattr,
                             const float* __restrict__ veff,
                             const float* __restrict__ asrc,
                             const float* __restrict__ adst,
                             float* __restrict__ abuf, float* __restrict__ mbuf) {
  int e = blockIdx.x * blockDim.x + threadIdx.x;
  if (e >= EALL) return;
  int s, d; const float* ea;
  if (e < N_EDGES) { s = ei[e]; d = ei[N_EDGES + e]; ea = eattr + (size_t)e * F_EDGE; }
  else             { s = d = e - N_EDGES;            ea = lattr + (size_t)s * F_EDGE; }
  float eav[F_EDGE];
  #pragma unroll
  for (int f = 0; f < F_EDGE; ++f) eav[f] = ea[f];
  #pragma unroll
  for (int h = 0; h < NHEAD; ++h) {
    float ae = 0.f;
    #pragma unroll
    for (int f = 0; f < F_EDGE; ++f) ae += eav[f] * veff[f * NHEAD + h];
    float al = asrc[s * NHEAD + h] + adst[d * NHEAD + h] + ae;
    al = (al > 0.f) ? al : 0.2f * al;          // leaky_relu 0.2
    abuf[(size_t)e * NHEAD + h] = al;
    atomicMaxF(&mbuf[d * NHEAD + h], al);
  }
}

__global__ void k_softmax_num(const int* __restrict__ ei,
                              float* __restrict__ abuf,
                              const float* __restrict__ mbuf,
                              float* __restrict__ dbuf) {
  int e = blockIdx.x * blockDim.x + threadIdx.x;
  if (e >= EALL) return;
  int d = (e < N_EDGES) ? ei[N_EDGES + e] : (e - N_EDGES);
  #pragma unroll
  for (int h = 0; h < NHEAD; ++h) {
    float ex = __expf(abuf[(size_t)e * NHEAD + h] - mbuf[d * NHEAD + h]);
    abuf[(size_t)e * NHEAD + h] = ex;
    atomicAdd(&dbuf[d * NHEAD + h], ex);
  }
}

__global__ void k_aggregate(const int* __restrict__ ei,
                            const float* __restrict__ abuf,
                            const float* __restrict__ dbuf,
                            const float* __restrict__ xs,
                            float* __restrict__ outacc) {
  size_t t = (size_t)blockIdx.x * blockDim.x + threadIdx.x;
  if (t >= (size_t)EALL * HIDD) return;
  int c = (int)(t & 127);
  int e = (int)(t >> 7);
  int s, d;
  if (e < N_EDGES) { s = ei[e]; d = ei[N_EDGES + e]; } else { s = d = e - N_EDGES; }
  int h = c >> 4;
  float w = abuf[(size_t)e * NHEAD + h] / (dbuf[d * NHEAD + h] + 1e-16f);
  atomicAdd(&outacc[(size_t)d * HIDD + c], w * xs[(size_t)s * HIDD + c]);
}

// ---------------- epilogues ----------------

__global__ void k_epilogue_cat(const float* __restrict__ outacc, const float* __restrict__ bias,
                               const float* __restrict__ g, const float* __restrict__ b,
                               const float* __restrict__ m, const float* __restrict__ v,
                               unsigned short* __restrict__ actbf) {
  int t = blockIdx.x * blockDim.x + threadIdx.x;
  if (t >= N_NODES * HIDD) return;
  int c = t & 127;
  float x = outacc[t] + bias[c];
  x = (x - m[c]) * rsqrtf(v[c] + EPS_BN) * g[c] + b[c];
  actbf[t] = f2bf(fmaxf(x, 0.f));
}

__global__ void k_epilogue_mean(const float* __restrict__ outacc, const float* __restrict__ bias,
                                const float* __restrict__ g, const float* __restrict__ b,
                                const float* __restrict__ m, const float* __restrict__ v,
                                float* __restrict__ h3) {
  int t = blockIdx.x * blockDim.x + threadIdx.x;
  if (t >= N_NODES * DHEAD) return;
  int n = t >> 4, dd = t & 15;
  float s = 0.f;
  #pragma unroll
  for (int h = 0; h < NHEAD; ++h) s += outacc[(size_t)n * HIDD + h * DHEAD + dd];
  s = s * (1.f / NHEAD) + bias[dd];
  h3[t] = (s - m[dd]) * rsqrtf(v[dd] + EPS_BN) * g[dd] + b[dd];
}

// ---------------- heads ----------------

__global__ void k_node_cls(const float* __restrict__ h3,
                           const float* __restrict__ W1, const float* __restrict__ b1,
                           const float* __restrict__ g, const float* __restrict__ bb,
                           const float* __restrict__ m, const float* __restrict__ v,
                           const float* __restrict__ W2, const float* __restrict__ b2,
                           float* __restrict__ out) {
  int n = blockIdx.x * blockDim.x + threadIdx.x;
  if (n >= N_NODES) return;
  float hv[DHEAD];
  #pragma unroll
  for (int k = 0; k < DHEAD; ++k) hv[k] = h3[(size_t)n * DHEAD + k];
  float t[32];
  for (int j = 0; j < 32; ++j) {
    float s = b1[j];
    #pragma unroll
    for (int k = 0; k < DHEAD; ++k) s += hv[k] * W1[k * 32 + j];
    s = fmaxf(s, 0.f);
    t[j] = (s - m[j]) * rsqrtf(v[j] + EPS_BN) * g[j] + bb[j];
  }
  #pragma unroll
  for (int o = 0; o < NFAULT; ++o) {
    float s = b2[o];
    for (int j = 0; j < 32; ++j) s += t[j] * W2[j * NFAULT + o];
    out[(size_t)n * NFAULT + o] = s;
  }
}

__global__ void k_pool(const float* __restrict__ h3, const int* __restrict__ batch,
                       float* __restrict__ gsum, float* __restrict__ gmax,
                       float* __restrict__ gcnt) {
  int t = blockIdx.x * blockDim.x + threadIdx.x;
  if (t >= N_NODES * DHEAD) return;
  int n = t >> 4, dd = t & 15;
  int bi = batch[n];
  float val = h3[t];
  atomicAdd(&gsum[bi * DHEAD + dd], val);
  atomicMaxF(&gmax[bi * DHEAD + dd], val);
  if (dd == 0) atomicAdd(&gcnt[bi], 1.f);
}

__global__ void k_graph_cls(const float* __restrict__ gsum, const float* __restrict__ gmax,
                            const float* __restrict__ gcnt,
                            const float* __restrict__ W1, const float* __restrict__ b1,
                            const float* __restrict__ g, const float* __restrict__ bb,
                            const float* __restrict__ m, const float* __restrict__ v,
                            const float* __restrict__ W2, const float* __restrict__ b2,
                            const float* __restrict__ W3, const float* __restrict__ b3,
                            float* __restrict__ out) {
  int gi = threadIdx.x;
  if (gi >= NBATCH) return;
  float gin[2 * DHEAD];
  float c = fmaxf(gcnt[gi], 1.f);
  #pragma unroll
  for (int d = 0; d < DHEAD; ++d) {
    gin[d] = gsum[gi * DHEAD + d] / c;
    gin[DHEAD + d] = gmax[gi * DHEAD + d];
  }
  float t1[64];
  for (int j = 0; j < 64; ++j) {
    float s = b1[j];
    for (int k = 0; k < 32; ++k) s += gin[k] * W1[k * 64 + j];
    s = fmaxf(s, 0.f);
    t1[j] = (s - m[j]) * rsqrtf(v[j] + EPS_BN) * g[j] + bb[j];
  }
  float t2[32];
  for (int j = 0; j < 32; ++j) {
    float s = b2[j];
    for (int k = 0; k < 64; ++k) s += t1[k] * W2[k * 32 + j];
    t2[j] = fmaxf(s, 0.f);
  }
  #pragma unroll
  for (int o = 0; o < NFAULT; ++o) {
    float s = b3[o];
    for (int k = 0; k < 32; ++k) s += t2[k] * W3[k * NFAULT + o];
    out[(size_t)N_NODES * NFAULT + gi * NFAULT + o] = s;
  }
}

// ---------------- host ----------------

#define GRID1(n) dim3((unsigned)(((size_t)(n) + 255) / 256)), dim3(256), 0, stream

extern "C" void kernel_launch(void* const* d_in, const int* in_sizes, int n_in,
                              void* d_out, int out_size, void* d_ws, size_t ws_size,
                              hipStream_t stream) {
  (void)in_sizes; (void)n_in; (void)out_size; (void)ws_size;
  const float* x     = (const float*)d_in[0];
  const int*   ei    = (const int*)d_in[1];
  const float* eattr = (const float*)d_in[2];
  const int*   batch = (const int*)d_in[3];

  const float *Wl[3]   = {(const float*)d_in[4],  (const float*)d_in[10], (const float*)d_in[16]};
  const float *Wel[3]  = {(const float*)d_in[5],  (const float*)d_in[11], (const float*)d_in[17]};
  const float *asl[3]  = {(const float*)d_in[6],  (const float*)d_in[12], (const float*)d_in[18]};
  const float *adl[3]  = {(const float*)d_in[7],  (const float*)d_in[13], (const float*)d_in[19]};
  const float *ael[3]  = {(const float*)d_in[8],  (const float*)d_in[14], (const float*)d_in[20]};
  const float *bl[3]   = {(const float*)d_in[9],  (const float*)d_in[15], (const float*)d_in[21]};
  const float *bng[3]  = {(const float*)d_in[22], (const float*)d_in[26], (const float*)d_in[30]};
  const float *bnb[3]  = {(const float*)d_in[23], (const float*)d_in[27], (const float*)d_in[31]};
  const float *bnm[3]  = {(const float*)d_in[24], (const float*)d_in[28], (const float*)d_in[32]};
  const float *bnv[3]  = {(const float*)d_in[25], (const float*)d_in[29], (const float*)d_in[33]};
  const float *nW1 = (const float*)d_in[34], *nb1 = (const float*)d_in[35];
  const float *nbg = (const float*)d_in[36], *nbb = (const float*)d_in[37];
  const float *nbm = (const float*)d_in[38], *nbv = (const float*)d_in[39];
  const float *nW2 = (const float*)d_in[40], *nb2 = (const float*)d_in[41];
  const float *gW1 = (const float*)d_in[42], *gb1 = (const float*)d_in[43];
  const float *gbg = (const float*)d_in[44], *gbb = (const float*)d_in[45];
  const float *gbm = (const float*)d_in[46], *gbv = (const float*)d_in[47];
  const float *gW2 = (const float*)d_in[48], *gb2 = (const float*)d_in[49];
  const float *gW3 = (const float*)d_in[50], *gb3 = (const float*)d_in[51];

  float* out = (float*)d_out;

  // workspace layout (256B aligned blocks)
  char* w = (char*)d_ws;
  size_t off = 0;
  auto take = [&](size_t bytes) -> char* {
    char* p = w + off;
    off += (bytes + 255) & ~(size_t)255;
    return p;
  };
  unsigned short* actbf = (unsigned short*)take((size_t)N_NODES * HIDD * 2);
  float* xs     = (float*)take((size_t)N_NODES * HIDD * 4);
  float* outacc = (float*)take((size_t)N_NODES * HIDD * 4);
  float* abuf   = (float*)take((size_t)EALL * NHEAD * 4);
  float* asrc   = (float*)take((size_t)N_NODES * NHEAD * 4);
  float* adst   = (float*)take((size_t)N_NODES * NHEAD * 4);
  float* mbuf   = (float*)take((size_t)N_NODES * NHEAD * 4);
  float* dbuf   = (float*)take((size_t)N_NODES * NHEAD * 4);
  float* lattr  = (float*)take((size_t)N_NODES * F_EDGE * 4);
  float* cnt    = (float*)take((size_t)N_NODES * 4);
  float* h3     = (float*)take((size_t)N_NODES * DHEAD * 4);
  unsigned int* Bp = (unsigned int*)take(8192 * 4);
  float* veff   = (float*)take(F_EDGE * NHEAD * 4);
  float* gsum   = (float*)take(NBATCH * DHEAD * 4);
  float* gmax   = (float*)take(NBATCH * DHEAD * 4);
  float* gcnt   = (float*)take(NBATCH * 4);

  // input activation -> bf16; self-loop edge attrs (shared by all layers)
  k_f32_to_bf16<<<GRID1(N_NODES * HIDD)>>>(x, actbf, N_NODES * HIDD);
  hipMemsetAsync(cnt, 0, (size_t)N_NODES * 4, stream);
  hipMemsetAsync(lattr, 0, (size_t)N_NODES * F_EDGE * 4, stream);
  k_loop_accum<<<GRID1(N_EDGES)>>>(ei, eattr, lattr, cnt);
  k_loop_div<<<GRID1(N_NODES * F_EDGE)>>>(lattr, cnt);

  for (int l = 0; l < 3; ++l) {
    k_pack_w<<<GRID1(8192)>>>(Wl[l], Bp);
    k_veff<<<dim3(1), dim3(256), 0, stream>>>(Wel[l], ael[l], veff);
    k_gemm_bf16<<<dim3(N_NODES / 16), dim3(256), 0, stream>>>(actbf, Bp, xs);
    k_asrc_adst<<<GRID1(N_NODES * NHEAD)>>>(xs, asl[l], adl[l], asrc, adst);
    k_fill_u32<<<GRID1(N_NODES * NHEAD)>>>((unsigned int*)mbuf, N_NODES * NHEAD, 0xFF800000u);
    hipMemsetAsync(dbuf, 0, (size_t)N_NODES * NHEAD * 4, stream);
    k_att_logits<<<GRID1(EALL)>>>(ei, eattr, lattr, veff, asrc, adst, abuf, mbuf);
    k_softmax_num<<<GRID1(EALL)>>>(ei, abuf, mbuf, dbuf);
    hipMemsetAsync(outacc, 0, (size_t)N_NODES * HIDD * 4, stream);
    k_aggregate<<<GRID1((size_t)EALL * HIDD)>>>(ei, abuf, dbuf, xs, outacc);
    if (l < 2)
      k_epilogue_cat<<<GRID1(N_NODES * HIDD)>>>(outacc, bl[l], bng[l], bnb[l], bnm[l], bnv[l], actbf);
    else
      k_epilogue_mean<<<GRID1(N_NODES * DHEAD)>>>(outacc, bl[2], bng[2], bnb[2], bnm[2], bnv[2], h3);
  }

  // node classifier
  k_node_cls<<<GRID1(N_NODES)>>>(h3, nW1, nb1, nbg, nbb, nbm, nbv, nW2, nb2, out);

  // graph pooling + classifier
  hipMemsetAsync(gsum, 0, NBATCH * DHEAD * 4, stream);
  hipMemsetAsync(gcnt, 0, NBATCH * 4, stream);
  k_fill_u32<<<dim3(1), dim3(256), 0, stream>>>((unsigned int*)gmax, NBATCH * DHEAD, 0xFF800000u);
  k_pool<<<GRID1(N_NODES * DHEAD)>>>(h3, batch, gsum, gmax, gcnt);
  k_graph_cls<<<dim3(1), dim3(64), 0, stream>>>(gsum, gmax, gcnt, gW1, gb1,
                                                gbg, gbb, gbm, gbv, gW2, gb2, gW3, gb3, out);
}
